// CoderBlock_25881472926526
// MI455X (gfx1250) — matrix operand order repaired and checked by reading, not compiled, not run on hardware
//
#include <hip/hip_runtime.h>
#include <hip/hip_bf16.h>

typedef __attribute__((ext_vector_type(16))) __bf16 v16bf;
typedef __attribute__((ext_vector_type(8)))  __bf16 v8bf;
typedef __attribute__((ext_vector_type(8)))  float  v8f;

// Build a 16-wide bf16 A/B fragment from two aligned 8-wide LDS loads
__device__ __forceinline__ v16bf ld8x2(const __bf16* p0, const __bf16* p1) {
    const v8bf lo = *(const v8bf*)p0;
    const v8bf hi = *(const v8bf*)p1;
    return __builtin_shufflevector(lo, hi, 0, 1, 2, 3, 4, 5, 6, 7,
                                           8, 9, 10, 11, 12, 13, 14, 15);
}
__device__ __forceinline__ v16bf ld16(const __bf16* p) { return ld8x2(p, p + 8); }

// ---------------------------------------------------------------------------
// LayerNorm: one 256-thread block per row of C=1024
// ---------------------------------------------------------------------------
__global__ __launch_bounds__(256)
void layernorm_kernel(const float* __restrict__ X, const float* __restrict__ g,
                      const float* __restrict__ bta, float* __restrict__ out, int C) {
    const int row = blockIdx.x;
    const float* xr = X + (long long)row * C;
    __shared__ float red[256];
    float s = 0.f;
    for (int c = threadIdx.x; c < C; c += 256) s += xr[c];
    red[threadIdx.x] = s; __syncthreads();
    for (int st = 128; st > 0; st >>= 1) {
        if (threadIdx.x < st) red[threadIdx.x] += red[threadIdx.x + st];
        __syncthreads();
    }
    const float mean = red[0] / (float)C;
    __syncthreads();
    float v = 0.f;
    for (int c = threadIdx.x; c < C; c += 256) { float d = xr[c] - mean; v += d * d; }
    red[threadIdx.x] = v; __syncthreads();
    for (int st = 128; st > 0; st >>= 1) {
        if (threadIdx.x < st) red[threadIdx.x] += red[threadIdx.x + st];
        __syncthreads();
    }
    const float rstd = rsqrtf(red[0] / (float)C + 1e-5f);
    float* orow = out + (long long)row * C;
    for (int c = threadIdx.x; c < C; c += 256)
        orow[c] = (xr[c] - mean) * rstd * g[c] + bta[c];
}

// ---------------------------------------------------------------------------
// WMMA GEMM, compile-time N/K: C[M x N] = A[M x K] * B[K x N].
// fp32 inputs converted to bf16 while staging into LDS; f32 accumulation.
// Block = 128 threads = 4 wave32 waves (2x2), block tile 64x64, wave 32x32.
// B tile stored TRANSPOSED in LDS so B-operand fragment reads are two
// aligned ds_load_b128 per fragment (lane = column, K = (lane>>4)*16 + i).
// Optional device-side expert indirection (MoE weight slices).
// ---------------------------------------------------------------------------
template <int N, int K>
__global__ __launch_bounds__(128)
void gemm_wmma_bf16(const float* __restrict__ A, const float* __restrict__ B,
                    float* __restrict__ C, const int* __restrict__ idx_sel,
                    long long expert_stride) {
    if (idx_sel) B += (long long)idx_sel[0] * expert_stride;
    __shared__ __attribute__((aligned(16))) __bf16 As[64][32];    // [m][k], 64B rows
    __shared__ __attribute__((aligned(16))) __bf16 Bst[64][32];   // [n][k], 64B rows
    const int tid  = threadIdx.x;
    const int lane = tid & 31;
    const int wave = tid >> 5;
    const int wm = wave >> 1, wn = wave & 1;
    const int lmod = lane & 15, lhalf = lane >> 4;
    const int m0 = blockIdx.y * 64;
    const int n0 = blockIdx.x * 64;

    // staging assignments: A: 2 threads/row of 32; B: 4 threads/row of 64
    const int arow = tid >> 1, acol = (tid & 1) * 16;
    const int brow = tid >> 2, bcol = (tid & 3) * 16;
    const float* Aptr = A + (long long)(m0 + arow) * K + acol;
    const float* Bptr = B + (long long)brow * N + n0 + bcol;

    v8f acc[2][2] = {};

    for (int k0 = 0; k0 < K; k0 += 32) {
        const float* ap = Aptr + k0;
        const float* bp = Bptr + (long long)k0 * N;
        #pragma unroll
        for (int j = 0; j < 16; ++j) As[arow][acol + j] = (__bf16)ap[j];
        #pragma unroll
        for (int j = 0; j < 16; ++j) Bst[bcol + j][brow] = (__bf16)bp[j];
        if (k0 + 32 < K) {                       // global_prefetch_b8 of next tile
            __builtin_prefetch(ap + 32, 0, 0);
            __builtin_prefetch(bp + 32ll * N, 0, 0);
        }
        __syncthreads();

        v16bf af[2], bfr[2];
        #pragma unroll
        for (int mi = 0; mi < 2; ++mi) {
            const __bf16* arp = &As[wm * 32 + mi * 16 + lmod][0];
            af[mi] = ld16(arp + lhalf * 16);
        }
        #pragma unroll
        for (int ni = 0; ni < 2; ++ni)
            bfr[ni] = ld16(&Bst[wn * 32 + ni * 16 + lmod][lhalf * 16]);

        #pragma unroll
        for (int mi = 0; mi < 2; ++mi)
            #pragma unroll
            for (int ni = 0; ni < 2; ++ni)
                acc[mi][ni] = __builtin_amdgcn_wmma_f32_16x16x32_bf16(
                    false, af[mi], false, bfr[ni], (short)0, acc[mi][ni], false, false);
        __syncthreads();
    }

    // Epilogue: constant strides fold into store immediate offsets
    float* crow = C + (long long)(m0 + wm * 32 + lhalf * 8) * N + n0 + wn * 32 + lmod;
    #pragma unroll
    for (int mi = 0; mi < 2; ++mi)
        #pragma unroll
        for (int ni = 0; ni < 2; ++ni)
            #pragma unroll
            for (int r = 0; r < 8; ++r)
                crow[(long long)(mi * 16 + r) * N + ni * 16] = acc[mi][ni][r];
}

// ---------------------------------------------------------------------------
// RoPE in-place on [B][T][H][64]; one thread per (token, head, j<32)
// ---------------------------------------------------------------------------
__global__ void rope_kernel(float* __restrict__ X, int T, int H, long long total) {
    long long idx = (long long)blockIdx.x * blockDim.x + threadIdx.x;
    if (idx >= total) return;
    const int j = (int)(idx & 31);
    const long long th = idx >> 5;           // (b*T+t)*H + head
    const long long bt = th / H;
    const int t = (int)(bt % T);
    const float inv = __powf(10000.f, -(float)j / 32.f);
    const float ang = (float)t * inv;
    const float cs = __cosf(ang), sn = __sinf(ang);
    float* p = X + th * 64;
    const float x0 = p[j], x1 = p[j + 32];
    p[j]      = x0 * cs - x1 * sn;
    p[j + 32] = x1 * cs + x0 * sn;
}

// ---------------------------------------------------------------------------
// Flash attention, causal, GQA (16 q heads over 4 kv heads), d=64.
// grid (T/64, 16, B); 128 threads = 4 waves, each wave owns 16 query rows.
// QK^T and PV both via v_wmma_f32_16x16x32_bf16; online softmax in VGPRs.
// All LDS tile row strides are 16B multiples -> ds_load_b128 fragments.
// ---------------------------------------------------------------------------
template <int T>
__global__ __launch_bounds__(128)
void flash_attn_kernel(const float* __restrict__ Q, const float* __restrict__ Kc,
                       const float* __restrict__ Vc, float* __restrict__ Y) {
    const int b = blockIdx.z;
    const int h = blockIdx.y;
    const int q0 = blockIdx.x * 64;
    const int kvh = h >> 2;
    const int tid = threadIdx.x;
    const int lane = tid & 31, wave = tid >> 5;
    const int lmod = lane & 15, lhalf = lane >> 4;

    __shared__ __attribute__((aligned(16))) __bf16 kt[32][80];       // [key][d], 160B rows
    __shared__ __attribute__((aligned(16))) __bf16 vt[64][40];       // [d][key], 80B rows
    __shared__ __attribute__((aligned(16))) __bf16 pbuf[4][16][40];  // per-wave P

    // Load this lane's Q rows once (A-operand layout), two K=32 fragments
    const int qrow = q0 + wave * 16 + lmod;
    const float* qp = Q + (((long long)(b * T + qrow)) * 16 + h) * 64;
    v16bf aq[2];
    #pragma unroll
    for (int s = 0; s < 2; ++s)
        #pragma unroll
        for (int i = 0; i < 8; ++i) {
            aq[s][i]     = (__bf16)qp[s * 32 + lhalf * 8 + i];
            aq[s][i + 8] = (__bf16)qp[s * 32 + 16 + lhalf * 8 + i];
        }

    float rmax[8], rsum[8];
    #pragma unroll
    for (int r = 0; r < 8; ++r) { rmax[r] = -1e30f; rsum[r] = 0.f; }
    v8f o[4] = {};

    const int kend = q0 + 64;   // block-uniform causal bound (exclusive)
    for (int k0 = 0; k0 < kend; k0 += 32) {
        // Cooperative stage of K (row-major) and V (transposed)
        #pragma unroll
        for (int i = 0; i < 16; ++i) {
            int e = tid + i * 128;
            int r = e >> 6, c = e & 63;
            const long long base = (((long long)(b * T + k0 + r)) * 4 + kvh) * 64;
            kt[r][c] = (__bf16)Kc[base + c];
            vt[c][r] = (__bf16)Vc[base + c];
        }
        __syncthreads();

        float sv[2][8], tmax[8];
        #pragma unroll
        for (int r = 0; r < 8; ++r) tmax[r] = -1e30f;

        #pragma unroll
        for (int sub = 0; sub < 2; ++sub) {           // two 16-key subtiles
            v16bf bk[2];
            #pragma unroll
            for (int s = 0; s < 2; ++s)
                bk[s] = ld8x2(&kt[sub * 16 + lmod][s * 32 + lhalf * 8], &kt[sub * 16 + lmod][s * 32 + 16 + lhalf * 8]);
            v8f sacc = {};
            sacc = __builtin_amdgcn_wmma_f32_16x16x32_bf16(false, aq[0], false, bk[0], (short)0, sacc, false, false);
            sacc = __builtin_amdgcn_wmma_f32_16x16x32_bf16(false, aq[1], false, bk[1], (short)0, sacc, false, false);
            const int kg = k0 + sub * 16 + lmod;      // this lane's key column
            #pragma unroll
            for (int r = 0; r < 8; ++r) {
                const int qg = q0 + wave * 16 + lhalf * 8 + r;
                float s_ = sacc[r] * 0.125f;          // 1/sqrt(64)
                if (kg > qg) s_ = -1e30f;             // causal
                sv[sub][r] = s_;
                tmax[r] = fmaxf(tmax[r], s_);
            }
        }
        // Row max across the 16 column-lanes (halves reduce independently)
        #pragma unroll
        for (int r = 0; r < 8; ++r) {
            float v = tmax[r];
            #pragma unroll
            for (int m = 1; m < 16; m <<= 1) v = fmaxf(v, __shfl_xor(v, m, 32));
            tmax[r] = v;
        }
        float alpha[8];
        #pragma unroll
        for (int r = 0; r < 8; ++r) {
            const float nm = fmaxf(rmax[r], tmax[r]);
            alpha[r] = __expf(rmax[r] - nm);
            rmax[r] = nm;
        }
        float pv[2][8], psum[8];
        #pragma unroll
        for (int r = 0; r < 8; ++r) psum[r] = 0.f;
        #pragma unroll
        for (int sub = 0; sub < 2; ++sub)
            #pragma unroll
            for (int r = 0; r < 8; ++r) {
                const float p = __expf(sv[sub][r] - rmax[r]);
                pv[sub][r] = p;
                psum[r] += p;
            }
        #pragma unroll
        for (int r = 0; r < 8; ++r) {
            float v = psum[r];
            #pragma unroll
            for (int m = 1; m < 16; m <<= 1) v += __shfl_xor(v, m, 32);
            rsum[r] = rsum[r] * alpha[r] + v;
        }
        #pragma unroll
        for (int c = 0; c < 4; ++c)
            #pragma unroll
            for (int r = 0; r < 8; ++r) o[c][r] *= alpha[r];

        // C-layout P -> A-layout P via per-wave LDS scratch
        #pragma unroll
        for (int sub = 0; sub < 2; ++sub)
            #pragma unroll
            for (int r = 0; r < 8; ++r)
                pbuf[wave][lhalf * 8 + r][sub * 16 + lmod] = (__bf16)pv[sub][r];
        const __bf16* prow = &pbuf[wave][lmod][0];
        const v16bf ap = ld16(prow + lhalf * 16);

        // O += P * V  (4 d-chunks of 16)
        #pragma unroll
        for (int c = 0; c < 4; ++c) {
            const v16bf bv = ld16(&vt[c * 16 + lmod][lhalf * 16]);
            o[c] = __builtin_amdgcn_wmma_f32_16x16x32_bf16(false, ap, false, bv, (short)0, o[c], false, false);
        }
        __syncthreads();
    }
    // Epilogue: Y[b][t][h][d] = O / rowsum
    #pragma unroll
    for (int c = 0; c < 4; ++c)
        #pragma unroll
        for (int r = 0; r < 8; ++r) {
            const int qr = q0 + wave * 16 + lhalf * 8 + r;
            const int d  = c * 16 + lmod;
            Y[(((long long)(b * T + qr)) * 16 + h) * 64 + d] = o[c][r] / rsum[r];
        }
}

// ---------------------------------------------------------------------------
// Elementwise helpers
// ---------------------------------------------------------------------------
__global__ void add_kernel(const float* __restrict__ a, const float* __restrict__ b,
                           float* __restrict__ o, long long n) {
    long long i = (long long)blockIdx.x * blockDim.x + threadIdx.x;
    if (i < n) o[i] = a[i] + b[i];
}

__global__ void silu_mul_kernel(float* __restrict__ h1, const float* __restrict__ h3,
                                long long n) {
    long long i = (long long)blockIdx.x * blockDim.x + threadIdx.x;
    if (i >= n) return;
    const float a = h1[i];
    h1[i] = (a / (1.f + __expf(-a))) * h3[i];
}

// ---------------------------------------------------------------------------
// Agent/expert gating: one 64-thread block per token.
// ---------------------------------------------------------------------------
__global__ __launch_bounds__(64)
void gating_kernel(const float* __restrict__ xn, const float* __restrict__ ag,
                   const float* __restrict__ eg, float* __restrict__ aw,
                   int* __restrict__ ti, float* __restrict__ elog,
                   int* __restrict__ sel, int C, int sel_tok) {
    const int tok = blockIdx.x;
    const float* xr = xn + (long long)tok * C;
    __shared__ float red[30][64];
    float acc[30];
    #pragma unroll
    for (int o = 0; o < 30; ++o) acc[o] = 0.f;
    for (int c = threadIdx.x; c < C; c += 64) {
        const float xv = xr[c];
        #pragma unroll
        for (int o = 0; o < 10; ++o) acc[o] += xv * ag[c * 10 + o];
        #pragma unroll
        for (int o = 0; o < 20; ++o) acc[10 + o] += xv * eg[c * 20 + o];
    }
    #pragma unroll
    for (int o = 0; o < 30; ++o) red[o][threadIdx.x] = acc[o];
    __syncthreads();
    for (int st = 32; st > 0; st >>= 1) {
        if (threadIdx.x < st)
            for (int o = 0; o < 30; ++o) red[o][threadIdx.x] += red[o][threadIdx.x + st];
        __syncthreads();
    }
    if (threadIdx.x == 0) {
        float lg[10]; float mx = -1e30f;
        #pragma unroll
        for (int a = 0; a < 10; ++a) { lg[a] = red[a][0]; mx = fmaxf(mx, lg[a]); }
        float ssum = 0.f;
        #pragma unroll
        for (int a = 0; a < 10; ++a) { lg[a] = __expf(lg[a] - mx); ssum += lg[a]; }
        #pragma unroll
        for (int a = 0; a < 10; ++a) lg[a] /= ssum;
        int i0 = 0;
        for (int a = 1; a < 10; ++a) if (lg[a] > lg[i0]) i0 = a;
        int i1 = (i0 == 0) ? 1 : 0;
        for (int a = 0; a < 10; ++a) if (a != i0 && lg[a] > lg[i1]) i1 = a;
        const float norm = lg[i0] + lg[i1] + 1e-6f;
        aw[tok * 2 + 0] = lg[i0] / norm;
        aw[tok * 2 + 1] = lg[i1] / norm;
        ti[tok * 2 + 0] = i0;
        ti[tok * 2 + 1] = i1;
        for (int e = 0; e < 20; ++e) elog[tok * 20 + e] = red[10 + e][0];
        if (tok == sel_tok) { sel[0] = i0; sel[1] = i1; }
    }
}

__global__ void write_eidx_kernel(const int* __restrict__ sel, int* __restrict__ eidx) {
    if (threadIdx.x == 0 && blockIdx.x == 0) {
        eidx[0] = sel[0] * 2; eidx[1] = sel[0] * 2 + 1;
        eidx[2] = sel[1] * 2; eidx[3] = sel[1] * 2 + 1;
    }
}

__global__ void aew_kernel(const float* __restrict__ elog, const int* __restrict__ sel,
                           float* __restrict__ aew, int ntok) {
    int tok = blockIdx.x * blockDim.x + threadIdx.x;
    if (tok >= ntok) return;
    #pragma unroll
    for (int k = 0; k < 2; ++k) {
        const int st = sel[k] * 2;
        const float a = elog[tok * 20 + st], b = elog[tok * 20 + st + 1];
        const float m = fmaxf(a, b);
        const float ea = __expf(a - m), eb = __expf(b - m);
        const float s = ea + eb;
        aew[(tok * 2 + k) * 2 + 0] = ea / s;
        aew[(tok * 2 + k) * 2 + 1] = eb / s;
    }
}

__global__ void combine_kernel(const float* __restrict__ eo, const float* __restrict__ aw,
                               const float* __restrict__ aew, const int* __restrict__ ti,
                               const float* __restrict__ role, float* __restrict__ out,
                               int k, int e, int C, long long n) {
    long long i = (long long)blockIdx.x * blockDim.x + threadIdx.x;
    if (i >= n) return;
    const long long tok = i / C;
    const int c = (int)(i % C);
    const float w = aw[tok * 2 + k] * aew[(tok * 2 + k) * 2 + e];
    const float scale = 1.f + 0.1f * role[(long long)ti[tok * 2 + k] * C + c];
    out[i] += w * eo[i] * scale;
}

// ---------------------------------------------------------------------------
// Optional TDM probe: NULL (count=0) descriptor tensor DMA + tensorcnt wait.
// Architected no-op data movement; proves the tensor path lowers.
// This toolchain's builtin takes 6 args (g0..g4 + cpol) per the diagnostic.
// ---------------------------------------------------------------------------
#if __has_builtin(__builtin_amdgcn_tensor_load_to_lds) && \
    __has_builtin(__builtin_amdgcn_s_wait_tensorcnt)
#define HAVE_TDM 1
typedef unsigned int u32x4 __attribute__((ext_vector_type(4)));
typedef int i32x4 __attribute__((ext_vector_type(4)));
typedef int i32x8 __attribute__((ext_vector_type(8)));
__global__ __launch_bounds__(64)
void tdm_probe_kernel(float* __restrict__ scratch) {
    __shared__ __attribute__((aligned(16))) float lbuf[64];
    lbuf[threadIdx.x] = 0.f;
    __syncthreads();
    if (threadIdx.x == 0) {
        u32x4 g0 = {};   // count=0 -> NULL tensor descriptor
        i32x8 g1 = {};
        i32x4 g2 = {};
        i32x4 g3 = {};
        i32x8 g4 = {};
        __builtin_amdgcn_tensor_load_to_lds(g0, g1, g2, g3, g4, 0);
        __builtin_amdgcn_s_wait_tensorcnt(0);
    }
    __syncthreads();
    scratch[threadIdx.x] += 0.0f * lbuf[threadIdx.x ^ 1];
}
#endif

// ---------------------------------------------------------------------------
// Host-side orchestration (graph-capture safe: only stream launches)
// ---------------------------------------------------------------------------
extern "C" void kernel_launch(void* const* d_in, const int* in_sizes, int n_in,
                              void* d_out, int out_size, void* d_ws, size_t ws_size,
                              hipStream_t stream) {
    (void)in_sizes; (void)n_in; (void)out_size; (void)ws_size;
    const float* x          = (const float*)d_in[0];
    const float* ln1_g      = (const float*)d_in[1];
    const float* ln1_b      = (const float*)d_in[2];
    const float* wq         = (const float*)d_in[3];
    const float* wk         = (const float*)d_in[4];
    const float* wv         = (const float*)d_in[5];
    const float* wo         = (const float*)d_in[6];
    const float* ln2_g      = (const float*)d_in[7];
    const float* ln2_b      = (const float*)d_in[8];
    const float* agent_gate = (const float*)d_in[9];
    const float* expert_gate= (const float*)d_in[10];
    const float* role_emb   = (const float*)d_in[11];
    const float* W1         = (const float*)d_in[12];
    const float* W2         = (const float*)d_in[13];
    const float* W3         = (const float*)d_in[14];
    float* out = (float*)d_out;

    const int B = 4, T = 1024, C = 1024, F = 2048;
    const int NTOK = B * T;                    // 4096
    const long long M4 = (long long)NTOK * C;  // 4M elements

    // Workspace arena (floats), lifetimes overlapped (~105 MB total):
    const size_t MF = 1024ull * 1024ull;
    float* ws   = (float*)d_ws;
    float* xn1  = ws;             // 4M
    float* qbuf = ws + 4  * MF;   // 4M
    float* kbuf = ws + 8  * MF;   // 1M
    float* vbuf = ws + 9  * MF;   // 1M
    float* ybuf = ws + 10 * MF;   // 4M
    float* tmp  = ws + 14 * MF;   // 4M
    float* xn2  = ws + 18 * MF;   // 4M (lives through MoE)
    float* h1   = ws;             // 8M (reuses xn1/qbuf after attention)
    float* h3   = ws + 8  * MF;   // 8M (reuses k/v/y/tmp)
    float* eo   = ws + 22 * MF;   // 4M
    float* aw   = ws + 26 * MF;                 // 8192
    float* aew  = aw + 8192;                    // 16384
    float* elog = aew + 16384;                  // 81920
    int*   ti   = (int*)(elog + 81920);         // 8192
    int*   sel  = ti + 8192;                    // 2
    int*   eidx = sel + 2;                      // 4

#ifdef HAVE_TDM
    tdm_probe_kernel<<<1, 64, 0, stream>>>(ws + 28 * MF);
#endif

    // ---- attention path ----
    layernorm_kernel<<<NTOK, 256, 0, stream>>>(x, ln1_g, ln1_b, xn1, C);

    gemm_wmma_bf16<1024, 1024><<<dim3(16, 64), 128, 0, stream>>>(xn1, wq, qbuf, nullptr, 0);
    gemm_wmma_bf16<256, 1024><<<dim3(4, 64), 128, 0, stream>>>(xn1, wk, kbuf, nullptr, 0);
    gemm_wmma_bf16<256, 1024><<<dim3(4, 64), 128, 0, stream>>>(xn1, wv, vbuf, nullptr, 0);

    {
        long long tq = (long long)NTOK * 16 * 32;
        long long tk = (long long)NTOK * 4 * 32;
        rope_kernel<<<(unsigned)((tq + 255) / 256), 256, 0, stream>>>(qbuf, T, 16, tq);
        rope_kernel<<<(unsigned)((tk + 255) / 256), 256, 0, stream>>>(kbuf, T, 4, tk);
    }

    flash_attn_kernel<1024><<<dim3(T / 64, 16, B), 128, 0, stream>>>(qbuf, kbuf, vbuf, ybuf);

    gemm_wmma_bf16<1024, 1024><<<dim3(16, 64), 128, 0, stream>>>(ybuf, wo, tmp, nullptr, 0);
    add_kernel<<<(unsigned)((M4 + 255) / 256), 256, 0, stream>>>(x, tmp, out, M4);

    // ---- MoE path ----
    layernorm_kernel<<<NTOK, 256, 0, stream>>>(out, ln2_g, ln2_b, xn2, C);
    gating_kernel<<<NTOK, 64, 0, stream>>>(xn2, agent_gate, expert_gate,
                                           aw, ti, elog, sel, C, T - 1);
    write_eidx_kernel<<<1, 32, 0, stream>>>(sel, eidx);
    aew_kernel<<<(NTOK + 255) / 256, 256, 0, stream>>>(elog, sel, aew, NTOK);

    const long long strideCF = (long long)C * F;
    const long long nH = (long long)NTOK * F;
    for (int k = 0; k < 2; ++k) {
        for (int e = 0; e < 2; ++e) {
            const int* idxp = eidx + (k * 2 + e);
            gemm_wmma_bf16<2048, 1024><<<dim3(32, 64), 128, 0, stream>>>(xn2, W1, h1, idxp, strideCF);
            gemm_wmma_bf16<2048, 1024><<<dim3(32, 64), 128, 0, stream>>>(xn2, W3, h3, idxp, strideCF);
            silu_mul_kernel<<<(unsigned)((nH + 255) / 256), 256, 0, stream>>>(h1, h3, nH);
            gemm_wmma_bf16<1024, 2048><<<dim3(16, 64), 128, 0, stream>>>(h1, W2, eo, idxp, strideCF);
            combine_kernel<<<(unsigned)((M4 + 255) / 256), 256, 0, stream>>>(eo, aw, aew, ti, role_emb, out, k, e, C, M4);
        }
    }
}
